// InstanceAggLayer_58815282152047
// MI455X (gfx1250) — compile-verified, matrix operand order checked by reading, not run on hardware
//
#include <hip/hip_runtime.h>
#include <hip/hip_bf16.h>

typedef _Float16 v16h __attribute__((ext_vector_type(16)));
typedef _Float16 v8h  __attribute__((ext_vector_type(8)));
typedef float    v8f  __attribute__((ext_vector_type(8)));
typedef float    v4f  __attribute__((ext_vector_type(4)));

union VH { v16h v; v8h h[2]; };

#define LRELU_ALPHA 0.2f

// -------------------------------------------------------------------------
// Kernel 1: f_X = feats_X @ P_X   (f32 in, f16 out into workspace)
// One block = 256 threads = 8 waves, each wave does a 16-row x 64-col tile.
// B = P^T staged in LDS as f16 with padded row stride (72 halves) so the
// per-lane ds_load_b128 fragment reads are bank-conflict free.
// -------------------------------------------------------------------------
__global__ __launch_bounds__(256)
void proj_kernel(const float* __restrict__ fT, const float* __restrict__ fP,
                 const float* __restrict__ fI,
                 const float* __restrict__ PT, const float* __restrict__ PP,
                 const float* __restrict__ PI,
                 _Float16* __restrict__ oT, _Float16* __restrict__ oP,
                 _Float16* __restrict__ oI, int NN)
{
    __shared__ _Float16 sB[64 * 72];   // P^T (n-major), padded stride 72 halves

    const float* feats; const float* P; _Float16* out;
    if (blockIdx.y == 0)      { feats = fT; P = PT; out = oT; }
    else if (blockIdx.y == 1) { feats = fP; P = PP; out = oP; }
    else                      { feats = fI; P = PI; out = oI; }

    const int t = threadIdx.x;
    #pragma unroll
    for (int i = 0; i < 16; ++i) {     // 4096 f32 -> f16 transposed
        int e = t + i * 256;
        int k = e >> 6, n = e & 63;
        sB[n * 72 + k] = (_Float16)P[e];
    }
    __syncthreads();

    const int L    = t & 31;
    const int wave = t >> 5;
    const int r0   = L & 15;           // row/col within 16-tile
    const int hi   = L >> 4;           // lane half select
    const int rowBase = blockIdx.x * 128 + wave * 16;

    int rowA = rowBase + r0;
    if (rowA > NN - 1) rowA = NN - 1;  // clamp gather for tail (stores masked)

    v8f acc[4] = {};

    #pragma unroll
    for (int s = 0; s < 2; ++s) {      // K = 64 = 2 x 32
        const int koff = 32 * s + 8 * hi;
        // A fragment: two contiguous 8-element f32 chunks -> f16
        const v4f* pa0 = (const v4f*)(feats + rowA * 64 + koff);
        const v4f* pa1 = (const v4f*)(feats + rowA * 64 + koff + 16);
        v4f a00 = pa0[0], a01 = pa0[1];
        v4f a10 = pa1[0], a11 = pa1[1];
        v8h aLo, aHi;
        #pragma unroll
        for (int j = 0; j < 4; ++j) {
            aLo[j]     = (_Float16)a00[j];
            aLo[j + 4] = (_Float16)a01[j];
            aHi[j]     = (_Float16)a10[j];
            aHi[j + 4] = (_Float16)a11[j];
        }
        VH a; a.h[0] = aLo; a.h[1] = aHi;

        #pragma unroll
        for (int nt = 0; nt < 4; ++nt) {
            const int n = nt * 16 + r0;
            VH b;
            b.h[0] = *(const v8h*)(&sB[n * 72 + koff]);
            b.h[1] = *(const v8h*)(&sB[n * 72 + koff + 16]);
            acc[nt] = __builtin_amdgcn_wmma_f32_16x16x32_f16(
                false, a.v, false, b.v, (short)0, acc[nt], false, false);
        }
    }

    // Store: wave-uniform fast path (no predication) for full tiles.
    if (rowBase + 16 <= NN) {
        #pragma unroll
        for (int nt = 0; nt < 4; ++nt) {
            #pragma unroll
            for (int v = 0; v < 8; ++v) {
                int row = rowBase + v + 8 * hi;
                int N   = nt * 16 + r0;
                out[row * 64 + N] = (_Float16)acc[nt][v];
            }
        }
    } else {
        #pragma unroll
        for (int nt = 0; nt < 4; ++nt) {
            #pragma unroll
            for (int v = 0; v < 8; ++v) {
                int row = rowBase + v + 8 * hi;
                int N   = nt * 16 + r0;
                if (row < NN) out[row * 64 + N] = (_Float16)acc[nt][v];
            }
        }
    }
}

// -------------------------------------------------------------------------
// Kernel 2: per-edge gather + [16,256]x[256,64] WMMA GEMM + LeakyReLU.
// A fragments are gathered straight from the f16 tables (L2-resident),
// laid out exactly per the 16-bit A-matrix VGPR spec:
//   lane L holds row L%16, K-chunks [8*(L/16),+8) and [16+8*(L/16),+8).
// W^T lives in LDS as f16 with padded stride 264 halves (conflict-free).
// Output (256 MB stream) is stored non-temporally so the 192 MB L2 keeps
// the 19 MB f16 gather tables + W resident.
// -------------------------------------------------------------------------
__global__ __launch_bounds__(256)
void edge_kernel(const _Float16* __restrict__ fT, const _Float16* __restrict__ fP,
                 const _Float16* __restrict__ fI,
                 const int* __restrict__ adj, const float* __restrict__ W,
                 float* __restrict__ out, int E)
{
    __shared__ _Float16 sW[64 * 264];  // W^T (n-major), padded stride 264 halves

    const int t = threadIdx.x;
    #pragma unroll
    for (int i = 0; i < 64; ++i) {     // 16384 f32 -> f16 transposed
        int e = t + i * 256;
        int k = e >> 6, n = e & 63;
        sW[n * 264 + k] = (_Float16)W[e];
    }
    __syncthreads();

    const int L    = t & 31;
    const int wave = t >> 5;
    const int r0   = L & 15;
    const int hi   = L >> 4;
    const int instBase = blockIdx.x * 128 + wave * 16;

    int inst = instBase + r0;
    if (inst > E - 1) inst = E - 1;    // clamp gather for tail (stores masked)
    const int i0 = __builtin_nontemporal_load(&adj[inst]);
    const int i1 = __builtin_nontemporal_load(&adj[E + inst]);
    const int i2 = __builtin_nontemporal_load(&adj[2 * E + inst]);
    const int i3 = __builtin_nontemporal_load(&adj[3 * E + inst]);

    v8f acc[4] = {};

    #pragma unroll
    for (int s = 0; s < 8; ++s) {      // K = 256 = 8 x 32
        const int seg = s >> 1;        // which 64-wide concat segment
        const _Float16* tab = (seg < 2) ? fT : (seg == 2 ? fP : fI);
        const int id = (seg == 0) ? i0 : (seg == 1) ? i1 : (seg == 2) ? i2 : i3;
        const int koff = (s & 1) * 32 + 8 * hi;   // within 64-wide segment
        const _Float16* src = tab + id * 64 + koff;
        VH a;
        a.h[0] = *(const v8h*)(src);          // global_load_b128 (L2 hit)
        a.h[1] = *(const v8h*)(src + 16);     // global_load_b128 (L2 hit)

        const int wko = 32 * s + 8 * hi;
        #pragma unroll
        for (int nt = 0; nt < 4; ++nt) {
            const int n = nt * 16 + r0;
            VH b;
            b.h[0] = *(const v8h*)(&sW[n * 264 + wko]);
            b.h[1] = *(const v8h*)(&sW[n * 264 + wko + 16]);
            acc[nt] = __builtin_amdgcn_wmma_f32_16x16x32_f16(
                false, a.v, false, b.v, (short)0, acc[nt], false, false);
        }
    }

    // Store: wave-uniform fast path, non-temporal streaming writes.
    if (instBase + 16 <= E) {
        #pragma unroll
        for (int nt = 0; nt < 4; ++nt) {
            #pragma unroll
            for (int v = 0; v < 8; ++v) {
                int row = instBase + v + 8 * hi;
                int N   = nt * 16 + r0;
                float x = acc[nt][v];
                x = fmaxf(x, LRELU_ALPHA * x);
                __builtin_nontemporal_store(x, &out[row * 64 + N]);
            }
        }
    } else {
        #pragma unroll
        for (int nt = 0; nt < 4; ++nt) {
            #pragma unroll
            for (int v = 0; v < 8; ++v) {
                int row = instBase + v + 8 * hi;
                int N   = nt * 16 + r0;
                float x = acc[nt][v];
                x = fmaxf(x, LRELU_ALPHA * x);
                if (row < E) __builtin_nontemporal_store(x, &out[row * 64 + N]);
            }
        }
    }
}

extern "C" void kernel_launch(void* const* d_in, const int* in_sizes, int n_in,
                              void* d_out, int out_size, void* d_ws, size_t ws_size,
                              hipStream_t stream) {
    const float* taxF = (const float*)d_in[0];
    const float* perF = (const float*)d_in[1];
    const float* itmF = (const float*)d_in[2];
    const int*   adj  = (const int*)d_in[3];
    const float* Pc   = (const float*)d_in[4];
    const float* Pp   = (const float*)d_in[5];
    const float* Pi   = (const float*)d_in[6];
    const float* W    = (const float*)d_in[7];
    float* out = (float*)d_out;

    const int NN = in_sizes[0] / 64;   // 50000 nodes
    const int E  = in_sizes[3] / 4;    // 1M edges

    _Float16* fws = (_Float16*)d_ws;   // 3 x NN x 64 f16 projected tables
    _Float16* oT = fws;
    _Float16* oP = fws + (size_t)NN * 64;
    _Float16* oI = fws + (size_t)NN * 64 * 2;

    dim3 blk(256);
    dim3 pg((NN + 127) / 128, 3);
    proj_kernel<<<pg, blk, 0, stream>>>(taxF, perF, itmF, Pc, Pp, Pi, oT, oP, oI, NN);

    dim3 eg((E + 127) / 128);
    edge_kernel<<<eg, blk, 0, stream>>>(oT, oP, oI, adj, W, out, E);
}
